// InterleavedDownstreamFork_64407329571548
// MI455X (gfx1250) — compile-verified, matrix operand order checked by reading, not run on hardware
//
#include <hip/hip_runtime.h>
#include <stdint.h>

// ---------------------------------------------------------------------------
// Problem constants (from reference): H=4096, L=8192, S=4096, B*T=8192
// ---------------------------------------------------------------------------
#define H_DIM 4096
#define L_DIM 8192
#define S_DIM 4096
#define BT_ROWS 8192

typedef __attribute__((ext_vector_type(16))) __bf16 v16bf;
typedef __attribute__((ext_vector_type(8)))  float  v8f;

__device__ __forceinline__ uint16_t f32_to_bf16(float f) {
  union { float f; uint32_t u; } c; c.f = f;
  uint32_t u = c.u;
  uint32_t r = u + 0x7FFFu + ((u >> 16) & 1u);   // round-to-nearest-even
  return (uint16_t)(r >> 16);
}

// gfx1250 async global->LDS DMA (ASYNCcnt-tracked, no VGPR round trip).
// %0 = LDS byte address (low 32 bits of flat shared address), %1 = global addr.
__device__ __forceinline__ void async_load_b128(uint32_t lds_addr, const void* gaddr) {
  asm volatile("global_load_async_to_lds_b128 %0, %1, off"
               :: "v"(lds_addr), "v"(gaddr)
               : "memory");
}
__device__ __forceinline__ void wait_async_all() {
  asm volatile("s_wait_asynccnt 0x0" ::: "memory");
}
__device__ __forceinline__ uint32_t lds_addr32(const void* p) {
  return (uint32_t)(uintptr_t)p;   // flat shared addr: low 32 bits = LDS offset
}

// ---------------------------------------------------------------------------
// Prep kernels: f32 -> bf16 conversions with fused block-scale dequant
// ---------------------------------------------------------------------------
__global__ __launch_bounds__(256)
void k_f32_to_bf16(const float* __restrict__ src, uint16_t* __restrict__ dst, int n) {
  int stride = gridDim.x * blockDim.x;
  for (int i = blockIdx.x * blockDim.x + threadIdx.x; i < n; i += stride)
    dst[i] = f32_to_bf16(src[i]);
}

// dst[o,i] = bf16( w[o,i] * scale[o/128, i/128] )
__global__ __launch_bounds__(256)
void k_scale_to_bf16(const float* __restrict__ w, const float* __restrict__ scale,
                     uint16_t* __restrict__ dst, int O, int I) {
  int n = O * I;
  int scols = I >> 7;
  int stride = gridDim.x * blockDim.x;
  for (int i = blockIdx.x * blockDim.x + threadIdx.x; i < n; i += stride) {
    int o  = i / I;
    int ii = i - o * I;
    float s = scale[(o >> 7) * scols + (ii >> 7)];
    dst[i] = f32_to_bf16(w[i] * s);
  }
}

// dst = bf16( k_w*k_scale + v_w*v_scale )  (folded K+V weight, valid since S==H)
__global__ __launch_bounds__(256)
void k_kv_combine(const float* __restrict__ kw, const float* __restrict__ ks,
                  const float* __restrict__ vw, const float* __restrict__ vs,
                  uint16_t* __restrict__ dst, int O, int I) {
  int n = O * I;
  int scols = I >> 7;
  int stride = gridDim.x * blockDim.x;
  for (int i = blockIdx.x * blockDim.x + threadIdx.x; i < n; i += stride) {
    int o  = i / I;
    int ii = i - o * I;
    int si = (o >> 7) * scols + (ii >> 7);
    dst[i] = f32_to_bf16(kw[i] * ks[si] + vw[i] * vs[si]);
  }
}

// ---------------------------------------------------------------------------
// bf16 WMMA GEMM:  C[m,n] = sum_k A[m,k] * B[n,k]   (both operands K-major)
// Block: 256 threads = 8 waves; C-tile 256x128; wave grid 4x2; wave tile 64x64
// (4x4 subtiles -> 16 WMMA per 16 ds_load_b128 per K=32 step).
// Global->LDS via async DMA, double buffered (fetch k+1 overlaps WMMA of k).
// MODE 0: C = bf16(relu(acc));  MODE 1: C = acc (f32);  MODE 2: C += acc (f32)
// ---------------------------------------------------------------------------
template<int MODE>
__global__ __launch_bounds__(256)
void k_gemm_bf16(const uint16_t* __restrict__ A,   // [M,K] bf16
                 const uint16_t* __restrict__ Bw,  // [N,K] bf16
                 void* __restrict__ Cout,
                 int M, int N, int K) {
  constexpr int TM = 256, TN = 128, TK = 64;
  constexpr int LDK   = TK + 8;                    // padded stride (halves)
  constexpr int BUFEA = TM * LDK;                  // A elements per tile buffer
  constexpr int BUFEB = TN * LDK;                  // B elements per tile buffer
  __shared__ uint16_t As[2 * BUFEA];               // 73728 B
  __shared__ uint16_t Bs[2 * BUFEB];               // 36864 B

  const int tid  = threadIdx.x;
  const int lane = tid & 31;
  const int wave = tid >> 5;
  const int wm   = wave >> 1;                      // 0..3  (64 rows each)
  const int wn   = wave & 1;                       // 0..1  (64 cols each)
  const int bm   = blockIdx.y * TM;
  const int bn   = blockIdx.x * TN;

  v8f acc[4][4];
  #pragma unroll
  for (int i = 0; i < 4; i++)
    #pragma unroll
    for (int j = 0; j < 4; j++)
      #pragma unroll
      for (int r = 0; r < 8; r++) acc[i][j][r] = 0.0f;

  // Per-thread DMA slots: 8 x 16B chunks of A tile + 4 of B tile.
  const int lrow = tid >> 3;                       // 0..31
  const int lcol = tid & 7;                        // 0..7 -> 16B chunk
  uint32_t ldsA[8], ldsB[4];
  const uint16_t* gA[8];
  const uint16_t* gB[4];
  #pragma unroll
  for (int p = 0; p < 8; p++) {
    const int row = lrow + p * 32;
    ldsA[p] = lds_addr32(&As[row * LDK + lcol * 8]);
    gA[p]   = A + (size_t)(bm + row) * K + lcol * 8;
  }
  #pragma unroll
  for (int p = 0; p < 4; p++) {
    const int row = lrow + p * 32;
    ldsB[p] = lds_addr32(&Bs[row * LDK + lcol * 8]);
    gB[p]   = Bw + (size_t)(bn + row) * K + lcol * 8;
  }

  // Prologue: issue tile 0 into buffer 0.
  #pragma unroll
  for (int p = 0; p < 8; p++) async_load_b128(ldsA[p], gA[p]);
  #pragma unroll
  for (int p = 0; p < 4; p++) async_load_b128(ldsB[p], gB[p]);

  const int NK = K / TK;
  for (int kt = 0; kt < NK; kt++) {
    const int cur = kt & 1;

    wait_async_all();                              // our batch for buf[cur] landed
    __syncthreads();                               // everyone's landed; buf[cur^1] free

    if (kt + 1 < NK) {                             // overlap fetch of next tile
      const uint32_t nbA = (uint32_t)((cur ^ 1) * BUFEA * 2);
      const uint32_t nbB = (uint32_t)((cur ^ 1) * BUFEB * 2);
      const int knext = (kt + 1) * TK;
      #pragma unroll
      for (int p = 0; p < 8; p++) async_load_b128(ldsA[p] + nbA, gA[p] + knext);
      #pragma unroll
      for (int p = 0; p < 4; p++) async_load_b128(ldsB[p] + nbB, gB[p] + knext);
    }

    const uint16_t* as = As + cur * BUFEA;
    const uint16_t* bs = Bs + cur * BUFEB;
    #pragma unroll
    for (int s = 0; s < 2; s++) {                  // two K=32 WMMA steps
      const int koff = s * 32 + (lane >> 4) * 16;  // lanes 0-15: K lo, 16-31: K hi
      const int rsel = lane & 15;
      v16bf afrag[4], bfrag[4];
      #pragma unroll
      for (int j = 0; j < 4; j++)
        bfrag[j] = *(const v16bf*)(bs + (wn * 64 + j * 16 + rsel) * LDK + koff);
      #pragma unroll
      for (int i = 0; i < 4; i++)
        afrag[i] = *(const v16bf*)(as + (wm * 64 + i * 16 + rsel) * LDK + koff);
      #pragma unroll
      for (int i = 0; i < 4; i++)
        #pragma unroll
        for (int j = 0; j < 4; j++)
          acc[i][j] = __builtin_amdgcn_wmma_f32_16x16x32_bf16(
              false, afrag[i], false, bfrag[j], (short)0, acc[i][j], false, false);
    }
  }

  // Epilogue. C/D layout: VGPR r, lane L -> m = r + 8*(L/16), n = L%16
  const int mrow0 = bm + wm * 64;
  const int ncol0 = bn + wn * 64;
  const int nlane = lane & 15;
  const int mhalf = (lane >> 4) * 8;
  #pragma unroll
  for (int i = 0; i < 4; i++) {
    #pragma unroll
    for (int j = 0; j < 4; j++) {
      const int n  = ncol0 + j * 16 + nlane;
      const int mb = mrow0 + i * 16 + mhalf;
      if (MODE == 0) {
        uint16_t* C = (uint16_t*)Cout;
        #pragma unroll
        for (int r = 0; r < 8; r++) {
          float v = acc[i][j][r];
          v = v > 0.0f ? v : 0.0f;                 // fused relu
          C[(size_t)(mb + r) * N + n] = f32_to_bf16(v);
        }
      } else if (MODE == 1) {
        float* C = (float*)Cout;
        #pragma unroll
        for (int r = 0; r < 8; r++)
          C[(size_t)(mb + r) * N + n] = acc[i][j][r];
      } else {
        float* C = (float*)Cout;
        #pragma unroll
        for (int r = 0; r < 8; r++) {
          size_t idx = (size_t)(mb + r) * N + n;
          C[idx] = C[idx] + acc[i][j][r];          // fuse k+v contribution
        }
      }
    }
  }
}

// ---------------------------------------------------------------------------
// Row LayerNorm over H=4096, one 256-thread block per row
// ---------------------------------------------------------------------------
__global__ __launch_bounds__(256)
void k_layernorm(const float* __restrict__ acc, const float* __restrict__ gamma,
                 const float* __restrict__ beta, float* __restrict__ out) {
  __shared__ float r1[256];
  __shared__ float r2[256];
  const int row = blockIdx.x;
  const int t   = threadIdx.x;
  const float* src = acc + (size_t)row * H_DIM;

  float vals[16];
  float lsum = 0.0f, lsq = 0.0f;
  #pragma unroll
  for (int i = 0; i < 4; i++) {
    int c = i * 256 + t;                           // 16B chunk id, 0..1023
    float4 v = *(const float4*)(src + c * 4);
    vals[i*4+0] = v.x; vals[i*4+1] = v.y; vals[i*4+2] = v.z; vals[i*4+3] = v.w;
    lsum += v.x + v.y + v.z + v.w;
    lsq  += v.x*v.x + v.y*v.y + v.z*v.z + v.w*v.w;
  }
  r1[t] = lsum; r2[t] = lsq;
  __syncthreads();
  for (int s = 128; s > 0; s >>= 1) {
    if (t < s) { r1[t] += r1[t + s]; r2[t] += r2[t + s]; }
    __syncthreads();
  }
  const float inv = 1.0f / (float)H_DIM;
  const float mu  = r1[0] * inv;
  const float var = r2[0] * inv - mu * mu;
  const float rstd = rsqrtf(var + 1e-5f);

  float* dst = out + (size_t)row * H_DIM;
  #pragma unroll
  for (int i = 0; i < 4; i++) {
    int c = i * 256 + t;
    float4 g = *(const float4*)(gamma + c * 4);
    float4 b = *(const float4*)(beta  + c * 4);
    float4 o;
    o.x = (vals[i*4+0] - mu) * rstd * g.x + b.x;
    o.y = (vals[i*4+1] - mu) * rstd * g.y + b.y;
    o.z = (vals[i*4+2] - mu) * rstd * g.z + b.z;
    o.w = (vals[i*4+3] - mu) * rstd * g.w + b.w;
    *(float4*)(dst + c * 4) = o;
  }
}

// ---------------------------------------------------------------------------
// Launch
// ---------------------------------------------------------------------------
extern "C" void kernel_launch(void* const* d_in, const int* in_sizes, int n_in,
                              void* d_out, int out_size, void* d_ws, size_t ws_size,
                              hipStream_t stream) {
  const float* x        = (const float*)d_in[0];
  const float* q_w      = (const float*)d_in[1];
  const float* q_scale  = (const float*)d_in[2];
  const float* k_w      = (const float*)d_in[3];
  const float* k_scale  = (const float*)d_in[4];
  const float* v_w      = (const float*)d_in[5];
  const float* v_scale  = (const float*)d_in[6];
  const float* q_down_w = (const float*)d_in[7];
  const float* gamma    = (const float*)d_in[8];
  const float* beta     = (const float*)d_in[9];
  float* out = (float*)d_out;

  char* ws = (char*)d_ws;
  const size_t MB = 1024ull * 1024ull;
  uint16_t* xb  = (uint16_t*)(ws + 0);            //  64 MiB: bf16 x        [8192,4096]
  uint16_t* wq  = (uint16_t*)(ws +  64 * MB);     //  64 MiB: bf16 q_w*s    [8192,4096]
  uint16_t* wkv = (uint16_t*)(ws + 128 * MB);     //  32 MiB: bf16 kw+vw    [4096,4096]
  uint16_t* wd  = (uint16_t*)(ws + 160 * MB);     //  64 MiB: bf16 q_down_w [4096,8192]
  uint16_t* q1  = (uint16_t*)(ws + 224 * MB);     // 128 MiB: bf16 relu(Q1) [8192,8192]
  float*    acc = (float*)   (ws + 352 * MB);     // 128 MiB: f32 accum     [8192,4096]

  // 1) precision/dequant prep
  k_f32_to_bf16  <<<2048, 256, 0, stream>>>(x, xb, BT_ROWS * H_DIM);
  k_scale_to_bf16<<<2048, 256, 0, stream>>>(q_w, q_scale, wq, L_DIM, H_DIM);
  k_kv_combine   <<<2048, 256, 0, stream>>>(k_w, k_scale, v_w, v_scale, wkv, S_DIM, H_DIM);
  k_f32_to_bf16  <<<2048, 256, 0, stream>>>(q_down_w, wd, H_DIM * L_DIM);

  // 2) Q1 = relu(x @ q_w^T) -> bf16
  k_gemm_bf16<0><<<dim3(L_DIM / 128, BT_ROWS / 256), 256, 0, stream>>>(
      xb, wq, q1, BT_ROWS, L_DIM, H_DIM);
  // 3) acc = Q1 @ q_down_w^T  (f32)
  k_gemm_bf16<1><<<dim3(H_DIM / 128, BT_ROWS / 256), 256, 0, stream>>>(
      q1, wd, acc, BT_ROWS, H_DIM, L_DIM);
  // 4) acc += x @ (k_w+v_w)^T  (fused k+v, valid since S==H)
  k_gemm_bf16<2><<<dim3(H_DIM / 128, BT_ROWS / 256), 256, 0, stream>>>(
      xb, wkv, acc, BT_ROWS, H_DIM, S_DIM);
  // 5) layernorm
  k_layernorm<<<BT_ROWS, 256, 0, stream>>>(acc, gamma, beta, out);
}
